// Attention_82841329205812
// MI455X (gfx1250) — matrix operand order repaired and checked by reading, not compiled, not run on hardware
//
#include <hip/hip_runtime.h>
#include <hip/hip_bf16.h>
#include <math.h>

typedef _Float16 v16h __attribute__((ext_vector_type(16)));
typedef _Float16 v8h  __attribute__((ext_vector_type(8)));
typedef _Float16 v4h  __attribute__((ext_vector_type(4)));
typedef float    v8f  __attribute__((ext_vector_type(8)));

constexpr int Bc = 2, Tc = 2048, Cc = 2048, Hc = 16, Dc = 128;
constexpr int Mc = Bc * Tc;  // 4096 rows for the projection GEMMs

// ---------------------------------------------------------------- WMMA helpers
static __device__ __forceinline__ v8f wmma16(v16h a, v16h b, v8f c) {
  // D = A(16x32 f16) * B(32x16 f16) + C(16x16 f32)
  return __builtin_amdgcn_wmma_f32_16x16x32_f16(false, a, false, b, (short)0, c,
                                                false, false);
}

// A-matrix fragment (16x32 f16): lane<16 holds K chunks {0..7,16..23},
// lane>=16 holds {8..15,24..31}; row = (lane&15) chosen by caller.
static __device__ __forceinline__ v16h frag_a(const _Float16* rb, int lane) {
  const int hi = (lane >> 4) & 1;
  union { v16h v; v8h h[2]; } u;
  u.h[0] = *(const v8h*)(rb + hi * 8);
  u.h[1] = *(const v8h*)(rb + 16 + hi * 8);
  return u.v;
}

// B-matrix fragment (32x16 f16): lane n holds column n; lane<16 holds K chunks
// {0..7,16..23}, lane>=16 holds {8..15,24..31} (same as frag_a).
static __device__ __forceinline__ v16h frag_b(const _Float16* rb, int lane) {
  const int hi = (lane >> 4) & 1;
  union { v16h v; v8h h[2]; } u;
  u.h[0] = *(const v8h*)(rb + hi * 8);
  u.h[1] = *(const v8h*)(rb + 16 + hi * 8);
  return u.v;
}

// ---------------------------------------------------------------- f32 -> f16
__global__ __launch_bounds__(256) void cvt_f32_to_f16(
    const float* __restrict__ in, _Float16* __restrict__ out, int n4) {
  int i = blockIdx.x * 256 + threadIdx.x;
  if (i >= n4) return;
  const float4 f = ((const float4*)in)[i];
  v4h h;
  h.x = (_Float16)f.x; h.y = (_Float16)f.y;
  h.z = (_Float16)f.z; h.w = (_Float16)f.w;
  ((v4h*)out)[i] = h;
}

// ---------------------------------------------------------------- GEMM y=A*W^T
// A:[M,K] f16 row-major, W:[N,K] f16 row-major (torch Linear weight layout).
// Block tile 256(M)x128(N), 8 waves each computing a 64x64 sub-tile
// (16 WMMA accumulators), K-step 32, double-buffered LDS with register
// prefetch so there is a single barrier per K-step.
// MODE 0: write f16 scattered to [B,H,T,D] (QKV).  MODE 1: write f32 [M,N].
template <int MODE>
__global__ __launch_bounds__(256) void gemm_awt(
    const _Float16* __restrict__ A, const _Float16* __restrict__ W,
    void* __restrict__ Out, int M, int N, int K) {
  constexpr int LS = 40;  // padded LDS row stride (halves), keeps 16B alignment
  __shared__ _Float16 lA[2][256 * LS];
  __shared__ _Float16 lB[2][128 * LS];
  const int tid = threadIdx.x;
  const int wave = tid >> 5, lane = tid & 31;
  const int ln = lane & 15, hi = (lane >> 4) & 1;
  const int m0 = blockIdx.y * 256, n0 = blockIdx.x * 128;
  const int wm = (wave & 3) * 64;   // wave's 64 M-rows
  const int wn = (wave >> 2) * 64;  // wave's 64 N-cols

  // Per-thread chunk coordinates (8 halves each). A: 1024 chunks, B: 512.
  const int arow = tid >> 2, akc = (tid & 3) * 8;

  v8h ra[4], rb[2];
  auto fetch = [&](int k0) {
#pragma unroll
    for (int c = 0; c < 4; ++c)
      ra[c] = *(const v8h*)&A[(size_t)(m0 + arow + c * 64) * K + k0 + akc];
#pragma unroll
    for (int c = 0; c < 2; ++c)
      rb[c] = *(const v8h*)&W[(size_t)(n0 + arow + c * 64) * K + k0 + akc];
  };
  auto stash = [&](int buf) {
#pragma unroll
    for (int c = 0; c < 4; ++c)
      *(v8h*)&lA[buf][(arow + c * 64) * LS + akc] = ra[c];
#pragma unroll
    for (int c = 0; c < 2; ++c)
      *(v8h*)&lB[buf][(arow + c * 64) * LS + akc] = rb[c];
  };

  v8f acc[4][4] = {};
  fetch(0);
  stash(0);
  __syncthreads();
  int cur = 0;
  for (int k0 = 0; k0 < K; k0 += 32) {
    const bool more = (k0 + 32) < K;  // uniform across the block
    if (more) fetch(k0 + 32);         // hide global latency behind WMMAs
    v16h af[4];
#pragma unroll
    for (int i = 0; i < 4; ++i)
      af[i] = frag_a(&lA[cur][(wm + i * 16 + ln) * LS], lane);
#pragma unroll
    for (int j = 0; j < 4; ++j) {
      v16h bf = frag_b(&lB[cur][(wn + j * 16 + ln) * LS], lane);
#pragma unroll
      for (int i = 0; i < 4; ++i) acc[i][j] = wmma16(af[i], bf, acc[i][j]);
    }
    if (more) stash(cur ^ 1);  // other buffer: safe while peers read cur
    __syncthreads();
    cur ^= 1;
  }

#pragma unroll
  for (int i = 0; i < 4; ++i)
#pragma unroll
    for (int j = 0; j < 4; ++j) {
      const int mb = m0 + wm + i * 16 + hi * 8;  // C/D layout: M = r (+8 hi)
      const int n = n0 + wn + j * 16 + ln;       // N = lane&15
#pragma unroll
      for (int r = 0; r < 8; ++r) {
        const int m = mb + r;
        const float v = acc[i][j][r];
        if (MODE == 0) {
          const int b = m >> 11, t = m & (Tc - 1);  // m = b*T + t
          const int h = n >> 7, d = n & (Dc - 1);   // n = h*D + d
          ((_Float16*)Out)[(((size_t)b * Hc + h) * Tc + t) * Dc + d] = (_Float16)v;
        } else {
          ((float*)Out)[(size_t)m * N + n] = v;
        }
      }
    }
}

// ---------------------------------------------------------------- flash attn
// Q,K,V: [B,H,T,D] f16.  O: [B,T,C] f16 (heads re-concatenated).
__global__ __launch_bounds__(256) void flash_attn(
    const _Float16* __restrict__ Q, const _Float16* __restrict__ K,
    const _Float16* __restrict__ V, _Float16* __restrict__ O) {
  constexpr int KS = 136;  // ldsK row stride (halves)
  constexpr int VS = 40;   // ldsVt / ldsP row stride (halves)
  __shared__ _Float16 lK[32 * KS];     // K tile, 32 keys x 128 d
  __shared__ _Float16 lVt[128 * VS];   // V tile transposed, 128 d x 32 keys
  __shared__ _Float16 lP[8][16 * VS];  // per-wave P tile, 16 q x 32 keys

  const int tid = threadIdx.x;
  const int wave = tid >> 5, lane = tid & 31;
  const int ln = lane & 15, hi = (lane >> 4) & 1;
  const int h = blockIdx.y, b = blockIdx.z;
  const size_t base = (size_t)(b * Hc + h) * Tc * Dc;
  const _Float16* Qb = Q + base;
  const _Float16* Kb = K + base;
  const _Float16* Vb = V + base;
  const int q0 = blockIdx.x * 128;
  const int qw = q0 + wave * 16;  // this wave's 16 query rows
  const float scale = 0.08838834764831845f;  // D^-0.5

  v16h qf[4];  // 16x128 Q fragment, K-split into 4x32
#pragma unroll
  for (int kk = 0; kk < 4; ++kk)
    qf[kk] = frag_a(&Qb[(size_t)(qw + ln) * Dc + kk * 32], lane);

  v8f o[8] = {};  // 16x128 f32 accumulator (8 d-chunks of 16)
  float mrow[8], lsum[8];
#pragma unroll
  for (int r = 0; r < 8; ++r) { mrow[r] = -INFINITY; lsum[r] = 0.f; }

  const int nkb = (q0 >> 5) + 4;  // causal: keys 0 .. q0+127
  for (int jb = 0; jb < nkb; ++jb) {
    const int kb = jb * 32;
    __syncthreads();
#pragma unroll
    for (int c = 0; c < 2; ++c) {  // K tile: 512 16B chunks
      int idx = tid + c * 256;
      int row = idx >> 4, dc = (idx & 15) * 8;
      *(v8h*)&lK[row * KS + dc] = *(const v8h*)&Kb[(size_t)(kb + row) * Dc + dc];
    }
#pragma unroll
    for (int c = 0; c < 2; ++c) {  // V tile: vector-load rows, scatter transpose
      int idx = tid + c * 256;
      int row = idx >> 4, d0 = (idx & 15) * 8;
      v8h vv = *(const v8h*)&Vb[(size_t)(kb + row) * Dc + d0];
#pragma unroll
      for (int e = 0; e < 8; ++e) lVt[(d0 + e) * VS + row] = vv[e];
    }
    __syncthreads();

    // S = Q * K^T  (16 q x 32 keys), f32
    v8f s0 = {}, s1 = {};
#pragma unroll
    for (int kk = 0; kk < 4; ++kk) {
      v16h b0 = frag_b(&lK[ln * KS + kk * 32], lane);
      v16h b1 = frag_b(&lK[(16 + ln) * KS + kk * 32], lane);
      s0 = wmma16(qf[kk], b0, s0);
      s1 = wmma16(qf[kk], b1, s1);
    }

    // online softmax; C/D element (M = r + 8*hi, N = ln) -> key = kb(+16) + ln
    _Float16* pw = &lP[wave][0];
#pragma unroll
    for (int r = 0; r < 8; ++r) {
      const int qrow = qw + r + hi * 8;
      float sa = (qrow >= kb + ln)      ? s0[r] * scale : -INFINITY;
      float sb = (qrow >= kb + 16 + ln) ? s1[r] * scale : -INFINITY;
      float mx = fmaxf(sa, sb);
#pragma unroll
      for (int off = 8; off; off >>= 1) mx = fmaxf(mx, __shfl_xor(mx, off, 32));
      const float mnew = fmaxf(mrow[r], mx);
      float alpha, p0, p1;
      if (mnew == -INFINITY) {  // fully-masked tile for this row
        alpha = 1.f; p0 = 0.f; p1 = 0.f;
      } else {
        alpha = __expf(mrow[r] - mnew);  // exp(-inf)=0 on first hit
        p0 = __expf(sa - mnew);
        p1 = __expf(sb - mnew);
      }
      mrow[r] = mnew;
      float ps = p0 + p1;
#pragma unroll
      for (int off = 8; off; off >>= 1) ps += __shfl_xor(ps, off, 32);
      lsum[r] = lsum[r] * alpha + ps;
#pragma unroll
      for (int t = 0; t < 8; ++t) o[t][r] *= alpha;
      _Float16* prow = &pw[(r + hi * 8) * VS];
      prow[ln] = (_Float16)p0;
      prow[16 + ln] = (_Float16)p1;
    }

    // same-wave LDS RAW across lanes: drain DS counter before re-reading P
    asm volatile("s_wait_dscnt 0" ::: "memory");
    v16h pf = frag_a(&pw[ln * VS], lane);  // P as 16x32 A-matrix
#pragma unroll
    for (int t = 0; t < 8; ++t) {          // O += P * V
      v16h vf = frag_b(&lVt[(t * 16 + ln) * VS], lane);
      o[t] = wmma16(pf, vf, o[t]);
    }
  }

  // normalize and write O[b, qrow, h*128 + d] (causal => lsum > 0)
#pragma unroll
  for (int t = 0; t < 8; ++t)
#pragma unroll
    for (int r = 0; r < 8; ++r) {
      const int qrow = qw + r + hi * 8;
      const int d = t * 16 + ln;
      const float val = o[t][r] / lsum[r];
      O[((size_t)b * Tc + qrow) * Cc + h * Dc + d] = (_Float16)val;
    }
}

// ---------------------------------------------------------------- launch
extern "C" void kernel_launch(void* const* d_in, const int* in_sizes, int n_in,
                              void* d_out, int out_size, void* d_ws,
                              size_t ws_size, hipStream_t stream) {
  (void)in_sizes; (void)n_in; (void)out_size; (void)ws_size;
  const float* x  = (const float*)d_in[0];
  // d_in[1] is the causal tril mask; computed analytically instead of loaded.
  const float* wq = (const float*)d_in[2];
  const float* wk = (const float*)d_in[3];
  const float* wv = (const float*)d_in[4];
  const float* wo = (const float*)d_in[5];

  const size_t BTC = (size_t)Bc * Tc * Cc;  // 8388608
  const size_t CC  = (size_t)Cc * Cc;       // 4194304
  auto align = [](size_t v) { return (v + 255) & ~(size_t)255; };
  char* p = (char*)d_ws;
  _Float16* xh  = (_Float16*)p; p += align(BTC * 2);
  _Float16* wqh = (_Float16*)p; p += align(CC * 2);
  _Float16* wkh = (_Float16*)p; p += align(CC * 2);
  _Float16* wvh = (_Float16*)p; p += align(CC * 2);
  _Float16* woh = (_Float16*)p; p += align(CC * 2);
  _Float16* qh  = (_Float16*)p; p += align(BTC * 2);
  _Float16* kh  = (_Float16*)p; p += align(BTC * 2);
  _Float16* vh  = (_Float16*)p; p += align(BTC * 2);
  _Float16* oh  = (_Float16*)p; p += align(BTC * 2);

  cvt_f32_to_f16<<<(int)(BTC / 4 / 256), 256, 0, stream>>>(x, xh, (int)(BTC / 4));
  cvt_f32_to_f16<<<(int)(CC / 4 / 256), 256, 0, stream>>>(wq, wqh, (int)(CC / 4));
  cvt_f32_to_f16<<<(int)(CC / 4 / 256), 256, 0, stream>>>(wk, wkh, (int)(CC / 4));
  cvt_f32_to_f16<<<(int)(CC / 4 / 256), 256, 0, stream>>>(wv, wvh, (int)(CC / 4));
  cvt_f32_to_f16<<<(int)(CC / 4 / 256), 256, 0, stream>>>(wo, woh, (int)(CC / 4));

  dim3 g(Cc / 128, Mc / 256);  // (16, 16) tiles of 256x128
  gemm_awt<0><<<g, 256, 0, stream>>>(xh, wqh, qh, Mc, Cc, Cc);
  gemm_awt<0><<<g, 256, 0, stream>>>(xh, wkh, kh, Mc, Cc, Cc);
  gemm_awt<0><<<g, 256, 0, stream>>>(xh, wvh, vh, Mc, Cc, Cc);

  dim3 ga(Tc / 128, Hc, Bc);  // (16, 16, 2)
  flash_attn<<<ga, 256, 0, stream>>>(qh, kh, vh, oh);

  gemm_awt<1><<<g, 256, 0, stream>>>(oh, woh, d_out, Mc, Cc, Cc);
}
